// W_MSA_83769042141359
// MI455X (gfx1250) — matrix operand order repaired and checked by reading, not compiled, not run on hardware
//
#include <hip/hip_runtime.h>
#include <hip/hip_bf16.h>

// ---------------------------------------------------------------------------
// W-MSA fused kernel for gfx1250 (MI455X).
// One workgroup (256 threads = 8 wave32) per 8x8 window. All GEMMs use
// v_wmma_f32_16x16x32_bf16 with f32 accumulation; operands staged in LDS.
// QKV is computed as one stacked [576x192]x[192x64] GEMM with A-fragment
// reuse across 4 N-tiles and an explicit A double buffer.
// ---------------------------------------------------------------------------

typedef __attribute__((ext_vector_type(16))) __bf16 v16bf;
typedef __attribute__((ext_vector_type(8)))  float  v8f;
typedef __attribute__((ext_vector_type(4)))  unsigned int v4u;

#define C_DIM   192
#define L_DIM   64
#define N_HEAD  6
#define DK      32
#define WS      8
#define SCALE_Q 0.17677669529663687f   // 1/sqrt(32)
#define WSZ     (C_DIM * C_DIM)        // one weight matrix, elements

__device__ __forceinline__ unsigned short f32_to_bf16(float f) {
    unsigned int u = __float_as_uint(f);
    u += 0x7fffu + ((u >> 16) & 1u);       // round-to-nearest-even
    return (unsigned short)(u >> 16);
}

union FragBF {
    v16bf v;
    v4u   q[2];
};

// A-matrix fragment (16x32 bf16, MxK), src row-major [m][k] with leading dim ld.
// ISA layout: lane L<16 -> row M=L, elems 0..7 = K {k0..k0+7}, elems 8..15 = {k0+16..k0+23}
//             lane L>=16 -> row M=L-16, K offset +8.
// Both 16B halves are 16B-aligned for every call site (ld and k0 multiples of 8).
__device__ __forceinline__ v16bf load_frag_a(const unsigned short* __restrict__ base,
                                             int ld, int m0, int k0) {
    const unsigned lane = threadIdx.x & 31u;
    const unsigned short* p = base + (unsigned)(m0 + (int)(lane & 15u)) * ld
                                   + k0 + (int)((lane >> 4u) << 3u);
    FragBF f;
    f.q[0] = *(const v4u*)(p);         // K = +0..7
    f.q[1] = *(const v4u*)(p + 16);    // K = +16..23
    return f.v;
}

// B-matrix fragment (32x16 bf16, KxN), src stored as [n][k] with leading dim ld
// (K contiguous per column). lane L<16 -> col N=L, elems 0..7 = K {k0..k0+7}, elems 8..15 = {k0+16..k0+23}; lane>=16: K+8.
__device__ __forceinline__ v16bf load_frag_b(const unsigned short* __restrict__ base,
                                             int ld, int n0, int k0) {
    const unsigned lane = threadIdx.x & 31u;
    const unsigned short* p = base + (unsigned)(n0 + (int)(lane & 15u)) * ld
                                   + k0 + (int)((lane >> 4u) << 3u);
    FragBF f;
    f.q[0] = *(const v4u*)(p);
    f.q[1] = *(const v4u*)(p + 16);
    return f.v;
}

__device__ __forceinline__ v8f wmma_bf16(v16bf a, v16bf b, v8f c) {
    return __builtin_amdgcn_wmma_f32_16x16x32_bf16(false, a, false, b, (short)0, c,
                                                   false, false);
}

// Convert the four 192x192 f32 weights to bf16 (stacked Wq|Wk|Wv|Wp) and build
// the concatenated QKV bias (576 floats) in workspace.
__global__ void __launch_bounds__(256) wmsa_cvt_weights(
        const float* __restrict__ Wq, const float* __restrict__ Wk,
        const float* __restrict__ Wv, const float* __restrict__ Wp,
        const float* __restrict__ bq, const float* __restrict__ bk,
        const float* __restrict__ bv,
        unsigned short* __restrict__ o) {
    int i = blockIdx.x * 256 + threadIdx.x;
    if (i < WSZ) {
        o[i]           = f32_to_bf16(Wq[i]);
        o[WSZ + i]     = f32_to_bf16(Wk[i]);
        o[2 * WSZ + i] = f32_to_bf16(Wv[i]);
        o[3 * WSZ + i] = f32_to_bf16(Wp[i]);
    }
    float* biasAll = (float*)(o + 4 * WSZ);        // 576 f32
    if (i < 3 * C_DIM) {
        float v = (i < C_DIM) ? bq[i] : (i < 2 * C_DIM) ? bk[i - C_DIM]
                                                        : bv[i - 2 * C_DIM];
        biasAll[i] = v;
    }
}

// LDS layout (bytes). Xls is reused as attention-output staging.
#define OFF_X   0                         // [64][192] bf16 (X, later attn out)
#define OFF_Q   24576                     // [64][192] bf16 (Q^T, pre-scaled)
#define OFF_K   49152                     // [64][192] bf16 (K^T)
#define OFF_V   73728                     // [192][64] bf16 (V, channel-major)
#define OFF_S   98304                     // [64][64] f32 (scores)
#define OFF_P   114688                    // [64][64] bf16 (probs)
#define LDS_BYTES 122880

__global__ void __launch_bounds__(256, 2) wmsa_kernel(
        const float* __restrict__ x,
        const unsigned short* __restrict__ Wqkv,   // stacked [576][192] bf16
        const unsigned short* __restrict__ Wpb,    // [192][192] bf16
        const float* __restrict__ biasAll,         // [576] f32 (bq|bk|bv)
        const float* __restrict__ bp,
        float* __restrict__ out) {
    extern __shared__ char smem[];
    unsigned short* Xls = (unsigned short*)(smem + OFF_X);
    unsigned short* Qt  = (unsigned short*)(smem + OFF_Q);
    unsigned short* Kt  = (unsigned short*)(smem + OFF_K);
    unsigned short* Vct = (unsigned short*)(smem + OFF_V);
    float*          Sls = (float*)        (smem + OFF_S);
    unsigned short* Pls = (unsigned short*)(smem + OFF_P);

    const int win = blockIdx.x;            // 4096 windows
    const int b  = win >> 10;
    const int wh = (win >> 5) & 31;
    const int wl = win & 31;
    const int tid  = threadIdx.x;
    // Provably wave-uniform wave index -> scalar strip math / scalar branches.
    const int wave = __builtin_amdgcn_readfirstlane(tid) >> 5;
    const int lane = tid & 31;
    const int lhi8 = (lane >> 4) << 3;     // +8 row offset for upper half-wave (C/D)

    const long xbase = (long)b * (C_DIM * 65536L) + (long)(wh * WS) * 256 + wl * WS;

    // ---- Stage X window into LDS as bf16, token-major [l][c] -------------
    for (int idx = tid; idx < (C_DIM * L_DIM) / 4; idx += 256) {
        int c  = idx >> 4;                 // channel
        int l  = (idx & 15) << 2;          // first of 4 tokens (same window row)
        int r  = l >> 3, s = l & 7;
        float4 v = *(const float4*)(x + xbase + (long)c * 65536 + r * 256 + s);
        Xls[(l + 0) * C_DIM + c] = f32_to_bf16(v.x);
        Xls[(l + 1) * C_DIM + c] = f32_to_bf16(v.y);
        Xls[(l + 2) * C_DIM + c] = f32_to_bf16(v.z);
        Xls[(l + 3) * C_DIM + c] = f32_to_bf16(v.w);
    }
    __syncthreads();

    // ---- Stacked QKV projection: [576x192] x [192x64] ---------------------
    // Strip = 1 M-tile x 4 N-tiles; A-fragment loaded once per k-step and
    // double-buffered so loads overlap the 4 WMMAs.
    for (int st = wave; st < 36; st += 8) {
        const int pr = (st >= 24) ? 2 : (st >= 12) ? 1 : 0;   // scalar
        const int tm = st - pr * 12;
        v8f acc[4] = {};
        v16bf a_cur = load_frag_a(Wqkv, C_DIM, st * 16, 0);
#pragma unroll
        for (int kt = 0; kt < 6; ++kt) {
            v16bf a_nxt = a_cur;
            if (kt < 5) a_nxt = load_frag_a(Wqkv, C_DIM, st * 16, (kt + 1) * 32);
#pragma unroll
            for (int tn = 0; tn < 4; ++tn) {
                v16bf bb = load_frag_b(Xls, C_DIM, tn * 16, kt * 32);
                acc[tn] = wmma_bf16(a_cur, bb, acc[tn]);
            }
            a_cur = a_nxt;
        }
#pragma unroll
        for (int tn = 0; tn < 4; ++tn) {
            int col = tn * 16 + (lane & 15);       // token
#pragma unroll
            for (int i = 0; i < 8; ++i) {
                int cg = st * 16 + i + lhi8;       // stacked output channel
                int c  = tm * 16 + i + lhi8;       // channel within projection
                float v = acc[tn][i] + biasAll[cg];
                if (pr == 0)       Qt[col * C_DIM + c] = f32_to_bf16(v * SCALE_Q);
                else if (pr == 1)  Kt[col * C_DIM + c] = f32_to_bf16(v);
                else               Vct[c * L_DIM + col] = f32_to_bf16(v);
            }
        }
    }
    __syncthreads();

    // ---- Attention per head ----------------------------------------------
    for (int hd = 0; hd < N_HEAD; ++hd) {
        // S = (Q_h)^T K_h : M=64 q-tokens, N=64 k-tokens, K=32
        for (int t = wave; t < 16; t += 8) {
            int tq = t >> 2, tk = t & 3;
            v16bf a  = load_frag_a(Qt, C_DIM, tq * 16, hd * DK);
            v16bf bb = load_frag_b(Kt, C_DIM, tk * 16, hd * DK);
            v8f acc = {};
            acc = wmma_bf16(a, bb, acc);
            int col = tk * 16 + (lane & 15);
#pragma unroll
            for (int i = 0; i < 8; ++i)
                Sls[(tq * 16 + i + lhi8) * L_DIM + col] = acc[i];
        }
        __syncthreads();

        // Row softmax over k (64 rows, one thread each; exp via v_exp_f32)
        if (tid < L_DIM) {
            const float* row = Sls + tid * L_DIM;
            float m = -3.4e38f;
#pragma unroll 8
            for (int j = 0; j < L_DIM; ++j) m = fmaxf(m, row[j]);
            float sum = 0.f;
#pragma unroll 8
            for (int j = 0; j < L_DIM; ++j) sum += __expf(row[j] - m);
            float inv = __frcp_rn(sum);
#pragma unroll 8
            for (int j = 0; j < L_DIM; ++j)
                Pls[tid * L_DIM + j] = f32_to_bf16(__expf(row[j] - m) * inv);
        }
        __syncthreads();

        // O_h = V_h * P^T : M=32 chans, N=64 q-tokens, K=64 k-tokens (2 steps)
        {
            int td = wave >> 2, tq = wave & 3;         // 8 tiles == 8 waves
            v8f acc = {};
#pragma unroll
            for (int kt = 0; kt < 2; ++kt) {
                v16bf a  = load_frag_a(Vct, L_DIM, hd * DK + td * 16, kt * 32);
                v16bf bb = load_frag_b(Pls, L_DIM, tq * 16, kt * 32);
                acc = wmma_bf16(a, bb, acc);
            }
            int col = tq * 16 + (lane & 15);           // q token
#pragma unroll
            for (int i = 0; i < 8; ++i) {
                int c = hd * DK + td * 16 + i + lhi8;  // merged-head channel
                Xls[col * C_DIM + c] = f32_to_bf16(acc[i]);   // X buffer reused
            }
        }
        __syncthreads();
    }

    // ---- Output projection Wp * Out + bp, scatter to global --------------
    // Half-strips: 1 M-tile x 2 N-tiles -> 24 strips, 3 per wave, 2x A reuse.
    for (int st = wave; st < 24; st += 8) {
        const int tm  = st >> 1;
        const int tn0 = (st & 1) << 1;                 // first of 2 N-tiles
        v8f acc[2] = {};
        v16bf a_cur = load_frag_a(Wpb, C_DIM, tm * 16, 0);
#pragma unroll
        for (int kt = 0; kt < 6; ++kt) {
            v16bf a_nxt = a_cur;
            if (kt < 5) a_nxt = load_frag_a(Wpb, C_DIM, tm * 16, (kt + 1) * 32);
#pragma unroll
            for (int j = 0; j < 2; ++j) {
                v16bf bb = load_frag_b(Xls, C_DIM, (tn0 + j) * 16, kt * 32);
                acc[j] = wmma_bf16(a_cur, bb, acc[j]);
            }
            a_cur = a_nxt;
        }
#pragma unroll
        for (int j = 0; j < 2; ++j) {
            int l = (tn0 + j) * 16 + (lane & 15);
            int r = l >> 3, s = l & 7;
#pragma unroll
            for (int i = 0; i < 8; ++i) {
                int c = tm * 16 + i + lhi8;
                out[xbase + (long)c * 65536 + r * 256 + s] = acc[j][i] + bp[c];
            }
        }
    }
}

extern "C" void kernel_launch(void* const* d_in, const int* in_sizes, int n_in,
                              void* d_out, int out_size, void* d_ws, size_t ws_size,
                              hipStream_t stream) {
    (void)in_sizes; (void)n_in; (void)out_size; (void)ws_size;
    const float* x  = (const float*)d_in[0];
    const float* Wq = (const float*)d_in[1];
    const float* bq = (const float*)d_in[2];
    const float* Wk = (const float*)d_in[3];
    const float* bk = (const float*)d_in[4];
    const float* Wv = (const float*)d_in[5];
    const float* bv = (const float*)d_in[6];
    const float* Wp = (const float*)d_in[7];
    const float* bp = (const float*)d_in[8];

    // Workspace: 4 x 192x192 bf16 (294912 B) + 576 f32 bias (2304 B)
    unsigned short* wbf = (unsigned short*)d_ws;

    wmsa_cvt_weights<<<(WSZ + 255) / 256, 256, 0, stream>>>(Wq, Wk, Wv, Wp,
                                                            bq, bk, bv, wbf);

    hipFuncSetAttribute(reinterpret_cast<const void*>(wmsa_kernel),
                        hipFuncAttributeMaxDynamicSharedMemorySize, LDS_BYTES);

    wmsa_kernel<<<4096, 256, LDS_BYTES, stream>>>(
        x,
        wbf,                    // stacked Wq|Wk|Wv
        wbf + 3 * WSZ,          // Wp
        (const float*)(wbf + 4 * WSZ),
        bp,
        (float*)d_out);
}
